// SelfAttention_74491912782338
// MI455X (gfx1250) — compile-verified, matrix operand order checked by reading, not run on hardware
//
#include <hip/hip_runtime.h>
#include <hip/hip_bf16.h>

// Problem constants (fixed by the reference)
#define BB 4
#define TT 2048
#define CC 1024
#define HH 16
#define HD 64

typedef __attribute__((ext_vector_type(16))) __bf16 v16bf;
typedef __attribute__((ext_vector_type(8)))  float  v8f;
typedef __attribute__((ext_vector_type(4)))  unsigned u32x4;
typedef __attribute__((ext_vector_type(8)))  int      i32x8;
typedef __attribute__((ext_vector_type(4)))  int      i32x4;

#if defined(__has_builtin)
#if __has_builtin(__builtin_amdgcn_tensor_load_to_lds) && \
    __has_builtin(__builtin_amdgcn_s_wait_tensorcnt)
#define HAVE_TDM 1
#endif
#endif

__device__ __forceinline__ unsigned short f2bf(float f) {
    union { float f; unsigned u; } v; v.f = f;
    unsigned r = v.u + 0x7FFFu + ((v.u >> 16) & 1u);   // round-to-nearest-even
    return (unsigned short)(r >> 16);
}

// 32-byte WMMA operand fragment: 16 bf16 = two 16B LDS/global loads
union Frag {
    v16bf v;
    uint4 q[2];
};

#ifdef HAVE_TDM
// 2D bf16 tile DMA: global (row stride = stride elements) -> LDS, rows compacted.
// D# per CDNA5 ISA 8.3/8.4: group0 {count=1, lds_addr, global_addr, type=2},
// group1 {data_size=2B, tensor_dim0=tile_w, tensor_dim1=tile_h, tile_dim0=tile_w,
// tile_dim1=tile_h, tensor_dim0_stride=stride}; groups 2/3 zero (<=2D tensor).
// This toolchain exposes the 6-arg builtin (g0, g1, g2, g3, g4, cpol).
__device__ __forceinline__ void tdm_load_2d_bf16(unsigned lds_addr, const void* g,
                                                 unsigned tile_w, unsigned tile_h,
                                                 unsigned long long stride)
{
    unsigned long long ga = (unsigned long long)g;
    u32x4 g0;
    g0[0] = 1u;                                            // count=1, user mode
    g0[1] = lds_addr;                                      // bits [63:32]
    g0[2] = (unsigned)ga;                                  // global_addr[31:0]
    g0[3] = (unsigned)((ga >> 32) & 0x01FFFFFFu)           // global_addr[56:32]
          | (2u << 30);                                    // type=2 ("image")
    i32x8 g1;
    g1[0] = (int)(1u << 16);                               // data_size=1 -> 2 bytes
    g1[1] = (int)((tile_w & 0xFFFFu) << 16);               // tensor_dim0[15:0]
    g1[2] = (int)(((tile_w >> 16) & 0xFFFFu)               // tensor_dim0[31:16]
          | ((tile_h & 0xFFFFu) << 16));                   // tensor_dim1[15:0]
    g1[3] = (int)(((tile_h >> 16) & 0xFFFFu)               // tensor_dim1[31:16]
          | ((tile_w & 0xFFFFu) << 16));                   // tile_dim0
    g1[4] = (int)(tile_h & 0xFFFFu);                       // tile_dim1 (tile_dim2=0)
    g1[5] = (int)(stride & 0xFFFFFFFFu);                   // tensor_dim0_stride[31:0]
    g1[6] = (int)((stride >> 32) & 0xFFFFu);               // tensor_dim0_stride[47:32]
    g1[7] = 0;
    i32x4 z4 = {0, 0, 0, 0};
    i32x8 z8 = {0, 0, 0, 0, 0, 0, 0, 0};
    __builtin_amdgcn_tensor_load_to_lds(g0, g1, z4, z4, z8, 0);
}
#endif

// -----------------------------------------------------------------------------
// GEMM: Out[M,N] = A[M,K] @ W[N,K]^T + bias[N]
// Block tile 64x128, 8 waves (2x4), wave tile 32x32, K-step 32 (bf16 WMMA).
// A_BF16: A is bf16 (TDM-staged) else fp32 (converted while staging to LDS).
// OUT_BF16: write bf16 (ushort) else fp32.
// -----------------------------------------------------------------------------
template <bool A_BF16, bool OUT_BF16>
__global__ void __launch_bounds__(256)
gemm_bias_wmma(const void* __restrict__ Ap, const float* __restrict__ W,
               const float* __restrict__ bias, void* __restrict__ Outp,
               int M, int N, int K)
{
    __shared__ unsigned short lA[64][32];    // 4 KB
    __shared__ unsigned short lB[128][32];   // 8 KB

    const int tid  = threadIdx.x;
    const int lane = tid & 31;
    const int wave = tid >> 5;
    const int mw   = wave & 1;               // 0..1
    const int nw   = wave >> 1;              // 0..3
    const int Mblk = blockIdx.x * 64;
    const int Nblk = blockIdx.y * 128;

    const int half = lane >> 4;              // 0 / 1
    const int l16  = lane & 15;
    const int klo  = half * 8;               // K offset of first 8 elements

    const float*          Af = (const float*)Ap;
    const unsigned short* Ab = (const unsigned short*)Ap;

#ifdef HAVE_TDM
    const unsigned lAaddr = (unsigned)(size_t)(void*)&lA[0][0];
#endif

    v8f acc[2][2] = {};

    for (int k0 = 0; k0 < K; k0 += 32) {
        __syncthreads();
        // ---- stage A tile 64x32 ----
        if (A_BF16) {
#ifdef HAVE_TDM
            if (wave == 0)
                tdm_load_2d_bf16(lAaddr, Ab + (size_t)Mblk * K + k0, 32, 64,
                                 (unsigned long long)K);
#else
            #pragma unroll
            for (int i = 0; i < 8; ++i) {
                int idx = tid + i * 256;
                int r = idx >> 5, c = idx & 31;
                lA[r][c] = Ab[(size_t)(Mblk + r) * K + k0 + c];
            }
#endif
        } else {
            #pragma unroll
            for (int i = 0; i < 8; ++i) {
                int idx = tid + i * 256;                 // 0..2047
                int r = idx >> 5, c = idx & 31;
                lA[r][c] = f2bf(Af[(size_t)(Mblk + r) * K + k0 + c]);
            }
        }
        // ---- stage B tile (W rows = output columns) 128x32 (16 / thread) ----
        #pragma unroll
        for (int i = 0; i < 16; ++i) {
            int idx = tid + i * 256;                 // 0..4095
            int r = idx >> 5, c = idx & 31;
            lB[r][c] = f2bf(W[(size_t)(Nblk + r) * K + k0 + c]);
        }
        if (k0 + 32 < K)                             // hint next K tile into cache
            __builtin_prefetch(&W[(size_t)(Nblk + (tid >> 1)) * K + k0 + 32], 0, 1);
#ifdef HAVE_TDM
        if (A_BF16 && wave == 0) __builtin_amdgcn_s_wait_tensorcnt(0);
#endif
        __syncthreads();

        // ---- fragments ----
        Frag a[2], b[2];
        #pragma unroll
        for (int mi = 0; mi < 2; ++mi) {
            const unsigned short* rp = &lA[mw * 32 + mi * 16 + l16][0];
            a[mi].q[0] = *(const uint4*)(rp + klo);
            a[mi].q[1] = *(const uint4*)(rp + 16 + klo);
        }
        #pragma unroll
        for (int ni = 0; ni < 2; ++ni) {
            const unsigned short* rp = &lB[nw * 32 + ni * 16 + l16][0];
            b[ni].q[0] = *(const uint4*)(rp + klo);
            b[ni].q[1] = *(const uint4*)(rp + 16 + klo);
        }
        #pragma unroll
        for (int mi = 0; mi < 2; ++mi)
            #pragma unroll
            for (int ni = 0; ni < 2; ++ni)
                acc[mi][ni] = __builtin_amdgcn_wmma_f32_16x16x32_bf16(
                    false, a[mi].v, false, b[ni].v, (short)0, acc[mi][ni],
                    false, false);
    }

    // ---- epilogue: bias + store ----
    #pragma unroll
    for (int mi = 0; mi < 2; ++mi) {
        #pragma unroll
        for (int ni = 0; ni < 2; ++ni) {
            int col = Nblk + nw * 32 + ni * 16 + l16;
            float bv = bias[col];
            #pragma unroll
            for (int r = 0; r < 8; ++r) {
                int row = Mblk + mw * 32 + mi * 16 + r + half * 8;
                float val = acc[mi][ni][r] + bv;
                if (OUT_BF16)
                    ((unsigned short*)Outp)[(size_t)row * N + col] = f2bf(val);
                else
                    ((float*)Outp)[(size_t)row * N + col] = val;
            }
        }
    }
}

// -----------------------------------------------------------------------------
// Causal flash attention, bf16 WMMA, f32 online softmax.
// grid = (T/64, B*H), block = 128 (4 waves); wave w owns queries q0+16w..+15.
// Per 32-key chunk: S = Q K^T (4 WMMA), online softmax, O += P V (4 WMMA).
// K chunk staged via TDM (tensor_load_to_lds); V transposed on the VALU path.
// -----------------------------------------------------------------------------
__global__ void __launch_bounds__(128)
flash_attn_wmma(const unsigned short* __restrict__ Q,
                const unsigned short* __restrict__ Kg,
                const unsigned short* __restrict__ Vg,
                unsigned short* __restrict__ Y)
{
    __shared__ unsigned short lK[32][64];      // key chunk, (key, hd) row-major, 4 KB
    __shared__ unsigned short lVt[64][32];     // V chunk transposed (hd, key), 4 KB
    __shared__ unsigned short lP[4][16][32];   // per-wave P bounce (C->A layout), 4 KB

    const int tid  = threadIdx.x;
    const int lane = tid & 31;
    const int wave = tid >> 5;
    const int half = lane >> 4;
    const int l16  = lane & 15;
    const int klo  = half * 8;

    const int q0   = blockIdx.x * 64;
    const int bh   = blockIdx.y;
    const int b    = bh / HH, h = bh % HH;
    const int row0 = b * TT;                   // token-row base in (B*T, C)
    const int col0 = h * HD;

#ifdef HAVE_TDM
    const unsigned lKaddr = (unsigned)(size_t)(void*)&lK[0][0];
#endif

    // Q fragments for this wave: 16 rows x 64 (two K=32 steps), straight from global
    const int qrow = row0 + q0 + wave * 16 + l16;
    Frag qf[2];
    #pragma unroll
    for (int kk = 0; kk < 2; ++kk) {
        const unsigned short* rp = Q + (size_t)qrow * CC + col0 + kk * 32;
        qf[kk].q[0] = *(const uint4*)(rp + klo);
        qf[kk].q[1] = *(const uint4*)(rp + 16 + klo);
    }

    v8f O[4] = {};
    v8f mrow, lrow;
    #pragma unroll
    for (int r = 0; r < 8; ++r) { mrow[r] = -1.0e30f; lrow[r] = 0.0f; }

    const float scale = 0.03125f;              // 1/sqrt(C) = 1/32 (reference scaling)
    const int   qbase = q0 + wave * 16 + half * 8;   // query row of element r is qbase+r
    const int   kend  = q0 + 64;

    for (int kc = 0; kc < kend; kc += 32) {
        __syncthreads();
        // ---- K chunk: 32 rows x 64 bf16, row stride C -> TDM DMA into lK ----
#ifdef HAVE_TDM
        if (wave == 0)
            tdm_load_2d_bf16(lKaddr, Kg + (size_t)(row0 + kc) * CC + col0,
                             64, 32, (unsigned long long)CC);
#endif
        // ---- V chunk staged transposed (1024 dwords, 8 / thread) ----
        #pragma unroll
        for (int i = 0; i < 8; ++i) {
            int idx = tid + i * 128;           // 0..1023 dword index
            int r = idx >> 5, c = (idx & 31) * 2;
            size_t g = (size_t)(row0 + kc + r) * CC + col0 + c;
#ifndef HAVE_TDM
            *(unsigned*)(&lK[r][c]) = *(const unsigned*)(Kg + g);
#endif
            unsigned uv = *(const unsigned*)(Vg + g);
            lVt[c][r]     = (unsigned short)(uv & 0xFFFFu);
            lVt[c + 1][r] = (unsigned short)(uv >> 16);
        }
#ifdef HAVE_TDM
        if (wave == 0) __builtin_amdgcn_s_wait_tensorcnt(0);
#endif
        __syncthreads();

        // ---- S = Q K^T : two 16-key tiles, HD=64 -> two K=32 WMMA steps ----
        v8f S[2];
        #pragma unroll
        for (int j = 0; j < 2; ++j) {
            v8f s = {};
            #pragma unroll
            for (int kk = 0; kk < 2; ++kk) {
                Frag bf;
                const unsigned short* rp = &lK[j * 16 + l16][kk * 32];
                bf.q[0] = *(const uint4*)(rp + klo);
                bf.q[1] = *(const uint4*)(rp + 16 + klo);
                s = __builtin_amdgcn_wmma_f32_16x16x32_bf16(
                        false, qf[kk].v, false, bf.v, (short)0, s, false, false);
            }
            S[j] = s;
        }

        // ---- scale + causal mask ----
        #pragma unroll
        for (int j = 0; j < 2; ++j) {
            int key = kc + j * 16 + l16;
            #pragma unroll
            for (int r = 0; r < 8; ++r) {
                float v = S[j][r] * scale;
                S[j][r] = (key > qbase + r) ? -1.0e30f : v;
            }
        }

        // ---- row max across the 16 lanes holding one row ----
        v8f mnew = mrow;
        #pragma unroll
        for (int r = 0; r < 8; ++r) mnew[r] = fmaxf(mnew[r], fmaxf(S[0][r], S[1][r]));
        #pragma unroll
        for (int off = 1; off < 16; off <<= 1)
            #pragma unroll
            for (int r = 0; r < 8; ++r)
                mnew[r] = fmaxf(mnew[r], __shfl_xor(mnew[r], off, 32));

        v8f alpha;
        #pragma unroll
        for (int r = 0; r < 8; ++r) alpha[r] = __expf(mrow[r] - mnew[r]);
        mrow = mnew;

        // ---- P = exp(S - m), row sum ----
        v8f psum = {};
        #pragma unroll
        for (int j = 0; j < 2; ++j)
            #pragma unroll
            for (int r = 0; r < 8; ++r) {
                float p = __expf(S[j][r] - mnew[r]);
                S[j][r] = p;
                psum[r] += p;
            }
        #pragma unroll
        for (int off = 1; off < 16; off <<= 1)
            #pragma unroll
            for (int r = 0; r < 8; ++r) psum[r] += __shfl_xor(psum[r], off, 32);
        #pragma unroll
        for (int r = 0; r < 8; ++r) lrow[r] = lrow[r] * alpha[r] + psum[r];

        // ---- C-layout -> A-layout via per-wave LDS bounce (same-wave DS is in-order)
        #pragma unroll
        for (int j = 0; j < 2; ++j)
            #pragma unroll
            for (int r = 0; r < 8; ++r)
                lP[wave][r + half * 8][j * 16 + l16] = f2bf(S[j][r]);

        Frag pf;
        {
            const unsigned short* rp = &lP[wave][l16][0];
            pf.q[0] = *(const uint4*)(rp + klo);
            pf.q[1] = *(const uint4*)(rp + 16 + klo);
        }

        // ---- O = O*alpha + P @ V over four 16-wide HD tiles ----
        #pragma unroll
        for (int n = 0; n < 4; ++n) {
            #pragma unroll
            for (int r = 0; r < 8; ++r) O[n][r] *= alpha[r];
            Frag vf;
            const unsigned short* rp = &lVt[n * 16 + l16][0];
            vf.q[0] = *(const uint4*)(rp + klo);
            vf.q[1] = *(const uint4*)(rp + 16 + klo);
            O[n] = __builtin_amdgcn_wmma_f32_16x16x32_bf16(
                       false, pf.v, false, vf.v, (short)0, O[n], false, false);
        }
    }

    // ---- epilogue: y = O / l, bf16 into workspace (input of final projection) ----
    #pragma unroll
    for (int n = 0; n < 4; ++n) {
        #pragma unroll
        for (int r = 0; r < 8; ++r) {
            int row = row0 + q0 + wave * 16 + r + half * 8;
            int col = col0 + n * 16 + l16;
            Y[(size_t)row * CC + col] = f2bf(O[n][r] / lrow[r]);
        }
    }
}

// -----------------------------------------------------------------------------
extern "C" void kernel_launch(void* const* d_in, const int* in_sizes, int n_in,
                              void* d_out, int out_size, void* d_ws, size_t ws_size,
                              hipStream_t stream)
{
    const float* x  = (const float*)d_in[0];
    const float* Wq = (const float*)d_in[1];
    const float* bq = (const float*)d_in[2];
    const float* Wk = (const float*)d_in[3];
    const float* bk = (const float*)d_in[4];
    const float* Wv = (const float*)d_in[5];
    const float* bv = (const float*)d_in[6];
    const float* Wo = (const float*)d_in[7];
    const float* bo = (const float*)d_in[8];
    float* out = (float*)d_out;

    const size_t nTok = (size_t)BB * TT;                    // 8192
    unsigned short* Qw = (unsigned short*)d_ws;             // 16 MB each
    unsigned short* Kw = Qw + nTok * CC;
    unsigned short* Vw = Kw + nTok * CC;
    unsigned short* Yw = Vw + nTok * CC;

    dim3 gblk((unsigned)(nTok / 64), CC / 128);             // 128 x 8
    gemm_bias_wmma<false, true><<<gblk, 256, 0, stream>>>(x, Wq, bq, Qw, (int)nTok, CC, CC);
    gemm_bias_wmma<false, true><<<gblk, 256, 0, stream>>>(x, Wk, bk, Kw, (int)nTok, CC, CC);
    gemm_bias_wmma<false, true><<<gblk, 256, 0, stream>>>(x, Wv, bv, Vw, (int)nTok, CC, CC);

    dim3 gat(TT / 64, BB * HH);                             // 32 x 64
    flash_attn_wmma<<<gat, 128, 0, stream>>>(Qw, Kw, Vw, Yw);

    gemm_bias_wmma<true, false><<<gblk, 256, 0, stream>>>(Yw, Wo, bo, out, (int)nTok, CC, CC);
}
